// MultiHeadAttentionPooling_76888504533412
// MI455X (gfx1250) — matrix operand order repaired and checked by reading, not compiled, not run on hardware
//
#include <hip/hip_runtime.h>
#include <stdint.h>

#define DD 256   // embedding dim
#define KK 128   // hidden dim per head
#define HH 4     // heads
#define GG 1024  // graphs

typedef __attribute__((ext_vector_type(16))) __bf16 v16bf;
typedef __attribute__((ext_vector_type(8)))  float  v8f;

union TileU {
  unsigned u[8];
  uint4    u4[2];
  v16bf    v;
};

__device__ inline unsigned pack_bf16(float lo, float hi) {
  unsigned bl = __float_as_uint(lo);
  unsigned bh = __float_as_uint(hi);
  bl = (bl + 0x7FFFu + ((bl >> 16) & 1u)) >> 16;   // round-to-nearest-even
  bh = (bh + 0x7FFFu + ((bh >> 16) & 1u)) >> 16;
  return (bh << 16) | (bl & 0xFFFFu);
}

// ---------------------------------------------------------------------------
// Kernel 1: repack W1 [H][D][K] fp32 -> bf16 pairs in WMMA B-operand order.
// B tile (h, kt, nt): 32 lanes x 8 dwords. lane L: column n = nt*16 + L%16,
// dword j: K = kt*32 + (L/16)*16 + 2j (+1 in high half).  (contiguous k order, same as the A slab)
// ---------------------------------------------------------------------------
__global__ __launch_bounds__(256) void prep_w1(const float* __restrict__ W1,
                                               unsigned* __restrict__ Bpk) {
  int t = blockIdx.x * blockDim.x + threadIdx.x;   // 0..65535
  int j    = t & 7;
  int lane = (t >> 3) & 31;
  int nt   = (t >> 8) & 7;
  int kt   = (t >> 11) & 7;
  int h    = (t >> 14) & 3;
  int n = nt * 16 + (lane & 15);
  int k = kt * 32 + (lane >> 4) * 16 + j * 2;
  const float* Wh = W1 + h * DD * KK;              // W1[h][d][kcol]
  float lo = Wh[k * KK + n];
  float hi = Wh[(k + 1) * KK + n];
  Bpk[t] = pack_bf16(lo, hi);
}

// ---------------------------------------------------------------------------
// Kernel 2: fused GEMM + bias + ReLU + W2-dot -> scores[N][H]
// Block = 256 threads = 8 waves; wave w owns nodes [base+16w, base+16w+16).
// A slab (16 nodes x 256 D, bf16) lives in 64 VGPRs for the whole kernel.
// B staged global->LDS asynchronously, double-buffered across heads (128KB).
// ---------------------------------------------------------------------------
__global__ __launch_bounds__(256) void score_kernel(
    const float* __restrict__ x, const unsigned* __restrict__ Bpk,
    const float* __restrict__ b1, const float* __restrict__ W2,
    const float* __restrict__ b2, float* __restrict__ scores, int n_nodes) {
  extern __shared__ unsigned char smem_raw[];
  unsigned* sB0 = (unsigned*)smem_raw;             // 64KB buffer 0
  unsigned* sB1 = sB0 + 16384;                     // 64KB buffer 1

  const int tid     = threadIdx.x;
  const int lane    = tid & 31;
  const int wave    = tid >> 5;
  const int m       = lane & 15;                   // A row within tile
  const int halfSel = lane >> 4;
  const int tileBase = blockIdx.x * 128 + wave * 16;

  // ---- load A slab, convert fp32 -> packed bf16 in WMMA A-operand order ----
  int node = tileBase + m;
  if (node >= n_nodes) node = n_nodes - 1;         // clamp; EXEC stays all-ones
  const float* xrow = x + (size_t)node * DD;

  TileU a[8];                                      // 8 kt tiles x 8 dwords = 64 VGPRs
#pragma unroll
  for (int kt = 0; kt < 8; ++kt) {
    int k0 = kt * 32 + halfSel * 16;
    float4 f0 = *(const float4*)(xrow + k0);
    float4 f1 = *(const float4*)(xrow + k0 + 4);
    float4 f2 = *(const float4*)(xrow + k0 + 8);
    float4 f3 = *(const float4*)(xrow + k0 + 12);
    a[kt].u[0] = pack_bf16(f0.x, f0.y);
    a[kt].u[1] = pack_bf16(f0.z, f0.w);
    a[kt].u[2] = pack_bf16(f1.x, f1.y);
    a[kt].u[3] = pack_bf16(f1.z, f1.w);
    a[kt].u[4] = pack_bf16(f2.x, f2.y);
    a[kt].u[5] = pack_bf16(f2.z, f2.w);
    a[kt].u[6] = pack_bf16(f3.x, f3.y);
    a[kt].u[7] = pack_bf16(f3.z, f3.w);
  }

  // issue async stage of head 0 into buffer 0
  {
    const unsigned* src = Bpk;
#pragma unroll
    for (int it = 0; it < 16; ++it) {              // 4096 b128 moves / 256 threads
      int flat = it * 256 + tid;
      unsigned lds_off = (unsigned)(uintptr_t)(sB0 + flat * 4);
      unsigned voff    = (unsigned)(flat * 16);
      asm volatile("global_load_async_to_lds_b128 %0, %1, %2"
                   :: "v"(lds_off), "v"(voff), "s"(src) : "memory");
    }
  }

  for (int h = 0; h < HH; ++h) {
    // ---- prefetch next head's B into the other LDS buffer ----
    if (h < HH - 1) {
      unsigned* dst = ((h + 1) & 1) ? sB1 : sB0;
      const unsigned* src = Bpk + (h + 1) * 16384;
#pragma unroll
      for (int it = 0; it < 16; ++it) {
        int flat = it * 256 + tid;
        unsigned lds_off = (unsigned)(uintptr_t)(dst + flat * 4);
        unsigned voff    = (unsigned)(flat * 16);
        asm volatile("global_load_async_to_lds_b128 %0, %1, %2"
                     :: "v"(lds_off), "v"(voff), "s"(src) : "memory");
      }
      // own async loads complete in order: <=16 outstanding means head h arrived
      asm volatile("s_wait_asynccnt 0x10" ::: "memory");
    } else {
      asm volatile("s_wait_asynccnt 0x0" ::: "memory");
    }
    __syncthreads();                               // all waves' stages visible

    unsigned* sB = (h & 1) ? sB1 : sB0;

    float sc[8] = {0.f, 0.f, 0.f, 0.f, 0.f, 0.f, 0.f, 0.f};
#pragma unroll
    for (int np = 0; np < 4; ++np) {               // process 2 column tiles/pass
      const int n0 = np * 2, n1 = np * 2 + 1;
      v8f acc0 = {0.f, 0.f, 0.f, 0.f, 0.f, 0.f, 0.f, 0.f};
      v8f acc1 = {0.f, 0.f, 0.f, 0.f, 0.f, 0.f, 0.f, 0.f};
      // preload kt=0 B tiles for both columns
      TileU b0, b1t;
      {
        const uint4* p0 = (const uint4*)(sB + ((0 * 8 + n0) * 32 + lane) * 8);
        const uint4* p1 = (const uint4*)(sB + ((0 * 8 + n1) * 32 + lane) * 8);
        b0.u4[0] = p0[0]; b0.u4[1] = p0[1];
        b1t.u4[0] = p1[0]; b1t.u4[1] = p1[1];
      }
#pragma unroll
      for (int kt = 0; kt < 8; ++kt) {             // reduction over D=256
        TileU c0, c1;
        if (kt < 7) {                              // software-pipelined B loads
          const uint4* p0 = (const uint4*)(sB + (((kt + 1) * 8 + n0) * 32 + lane) * 8);
          const uint4* p1 = (const uint4*)(sB + (((kt + 1) * 8 + n1) * 32 + lane) * 8);
          c0.u4[0] = p0[0]; c0.u4[1] = p0[1];
          c1.u4[0] = p1[0]; c1.u4[1] = p1[1];
        }
        acc0 = __builtin_amdgcn_wmma_f32_16x16x32_bf16(
            false, a[kt].v, false, b0.v, (short)0, acc0, false, false);
        acc1 = __builtin_amdgcn_wmma_f32_16x16x32_bf16(
            false, a[kt].v, false, b1t.v, (short)0, acc1, false, false);
        if (kt < 7) { b0 = c0; b1t = c1; }
      }
      // epilogue: bias + ReLU + multiply by W2 column weight, accumulate score
      int kk0 = n0 * 16 + m;
      int kk1 = n1 * 16 + m;
      float b1v0 = b1[h * KK + kk0], w2v0 = W2[h * KK + kk0];
      float b1v1 = b1[h * KK + kk1], w2v1 = W2[h * KK + kk1];
#pragma unroll
      for (int j = 0; j < 8; ++j) {
        float t0 = acc0[j] + b1v0;
        float t1 = acc1[j] + b1v1;
        t0 = t0 > 0.f ? t0 : 0.f;
        t1 = t1 > 0.f ? t1 : 0.f;
        sc[j] += t0 * w2v0 + t1 * w2v1;
      }
    }
    // reduce over the 16 k-columns held across lanes (within each 16-lane half)
#pragma unroll
    for (int mask = 1; mask < 16; mask <<= 1) {
#pragma unroll
      for (int j = 0; j < 8; ++j) sc[j] += __shfl_xor(sc[j], mask, 32);
    }
    if (m == 0) {
      float bias = b2[h];
#pragma unroll
      for (int j = 0; j < 8; ++j) {
        int nd = tileBase + halfSel * 8 + j;       // C-layout: M = j + 8*halfSel
        if (nd < n_nodes) scores[(size_t)nd * HH + h] = sc[j] + bias;
      }
    }
    __syncthreads();                               // buffer reused in 2 heads
  }
}

// ---------------------------------------------------------------------------
__device__ inline int lower_bound_seg(const int* seg, int n, int val) {
  int lo = 0, hi = n;
  while (lo < hi) {
    int mid = (lo + hi) >> 1;
    if (seg[mid] < val) lo = mid + 1; else hi = mid;
  }
  return lo;
}

// Kernel 3: per-graph, per-head max and sum(exp) of scores.
__global__ __launch_bounds__(256) void stats_kernel(
    const float* __restrict__ scores, const int* __restrict__ seg,
    float* __restrict__ stats, int n_nodes) {
  __shared__ float red[256];
  __shared__ float smax[HH];
  int g = blockIdx.x, tid = threadIdx.x;
  int lo = lower_bound_seg(seg, n_nodes, g);
  int hi = lower_bound_seg(seg, n_nodes, g + 1);

  float mx[HH] = {-3.0e38f, -3.0e38f, -3.0e38f, -3.0e38f};
  for (int i = lo + tid; i < hi; i += 256) {
#pragma unroll
    for (int h = 0; h < HH; ++h) mx[h] = fmaxf(mx[h], scores[(size_t)i * HH + h]);
  }
#pragma unroll
  for (int h = 0; h < HH; ++h) {
    red[tid] = mx[h]; __syncthreads();
    for (int s = 128; s > 0; s >>= 1) {
      if (tid < s) red[tid] = fmaxf(red[tid], red[tid + s]);
      __syncthreads();
    }
    if (tid == 0) smax[h] = red[0];
    __syncthreads();
  }
  float sm[HH] = {0.f, 0.f, 0.f, 0.f};
  for (int i = lo + tid; i < hi; i += 256) {
#pragma unroll
    for (int h = 0; h < HH; ++h) sm[h] += __expf(scores[(size_t)i * HH + h] - smax[h]);
  }
#pragma unroll
  for (int h = 0; h < HH; ++h) {
    red[tid] = sm[h]; __syncthreads();
    for (int s = 128; s > 0; s >>= 1) {
      if (tid < s) red[tid] += red[tid + s];
      __syncthreads();
    }
    if (tid == 0) { stats[g * 8 + h] = smax[h]; stats[g * 8 + 4 + h] = red[0]; }
    __syncthreads();
  }
}

// Kernel 4: per-node mean-over-heads attention weight.
__global__ __launch_bounds__(256) void weight_kernel(
    const float* __restrict__ scores, const int* __restrict__ seg,
    const float* __restrict__ stats, float* __restrict__ wbuf, int n_nodes) {
  int i = blockIdx.x * blockDim.x + threadIdx.x;
  if (i >= n_nodes) return;
  int g = seg[i];
  float w = 0.f;
#pragma unroll
  for (int h = 0; h < HH; ++h) {
    float e = __expf(scores[(size_t)i * HH + h] - stats[g * 8 + h]);
    w += e / stats[g * 8 + 4 + h];
  }
  wbuf[i] = 0.25f * w;
}

// Kernel 5: weighted segment sum.  Block = graph, thread = output column.
__global__ __launch_bounds__(256) void pool_kernel(
    const float* __restrict__ x, const float* __restrict__ wbuf,
    const int* __restrict__ seg, float* __restrict__ out, int n_nodes) {
  int g = blockIdx.x, tid = threadIdx.x;
  int lo = lower_bound_seg(seg, n_nodes, g);
  int hi = lower_bound_seg(seg, n_nodes, g + 1);
  float a0 = 0.f, a1 = 0.f, a2 = 0.f, a3 = 0.f;
  int i = lo;
  for (; i + 3 < hi; i += 4) {                     // 4 rows in flight
    a0 += wbuf[i]     * x[(size_t)(i)     * DD + tid];
    a1 += wbuf[i + 1] * x[(size_t)(i + 1) * DD + tid];
    a2 += wbuf[i + 2] * x[(size_t)(i + 2) * DD + tid];
    a3 += wbuf[i + 3] * x[(size_t)(i + 3) * DD + tid];
  }
  for (; i < hi; ++i) a0 += wbuf[i] * x[(size_t)i * DD + tid];
  out[(size_t)g * DD + tid] = (a0 + a1) + (a2 + a3);
}

// ---------------------------------------------------------------------------
extern "C" void kernel_launch(void* const* d_in, const int* in_sizes, int n_in,
                              void* d_out, int out_size, void* d_ws, size_t ws_size,
                              hipStream_t stream) {
  const float* x   = (const float*)d_in[0];
  const int*   seg = (const int*)d_in[1];
  // d_in[2] = num_graphs scalar (G fixed at 1024)
  const float* W1  = (const float*)d_in[3];
  const float* b1  = (const float*)d_in[4];
  const float* W2  = (const float*)d_in[5];
  const float* b2  = (const float*)d_in[6];
  float* out = (float*)d_out;
  const int n = in_sizes[1];                      // N nodes

  // workspace layout (dwords): Bpk[65536] | scores[N*4] | stats[G*8] | w[N]
  unsigned* Bpk   = (unsigned*)d_ws;
  float* scores   = (float*)d_ws + 65536;
  float* stats    = scores + (size_t)n * HH;
  float* wbuf     = stats + GG * 8;

  prep_w1<<<256, 256, 0, stream>>>(W1, Bpk);
  int nblocks = (n + 127) / 128;
  score_kernel<<<nblocks, 256, 131072, stream>>>(x, Bpk, b1, W2, b2, scores, n);
  stats_kernel<<<GG, 256, 0, stream>>>(scores, seg, stats, n);
  weight_kernel<<<(n + 255) / 256, 256, 0, stream>>>(scores, seg, stats, wbuf, n);
  pool_kernel<<<GG, 256, 0, stream>>>(x, wbuf, seg, out, n);
}
